// myGNN_42460046688362
// MI455X (gfx1250) — matrix operand order repaired and checked by reading, not compiled, not run on hardware
//
#include <hip/hip_runtime.h>
#include <hip/hip_bf16.h>
#include <math.h>

// ---------------------------------------------------------------------------
// GAT-v2 GNN for MI455X (gfx1250, wave32).
//  - Node GEMMs (enc, Wl, Wr per layer): bf16 WMMA 16x16x32, f32 accumulate.
//  - Edge phase: 3 passes (logits+segment-max, exp+segment-sum, alpha-scatter),
//    float4 gathers, atomics accumulate directly into the residual buffer h.
//  - LN+ReLU fused with f32->bf16 conversion feeding the WMMA GEMMs.
// ---------------------------------------------------------------------------

#define NNODES 100000
#define NEDGES 1600000
#define DIM    128      // IN_DIM = IN_CH = HEADS*HID
#define HEADS  4
#define HID    32
#define NLAYER 3

typedef __attribute__((ext_vector_type(16))) __bf16 v16bf;
typedef __attribute__((ext_vector_type(8)))  __bf16 v8bf;
typedef __attribute__((ext_vector_type(4)))  __bf16 v4bf;
typedef __attribute__((ext_vector_type(8)))  float  v8f;

// ---------------- helpers ----------------

__device__ __forceinline__ float lrelu02(float v) { return v > 0.0f ? v : 0.2f * v; }

__device__ __forceinline__ float waveReduceSum(float v) {
#pragma unroll
  for (int m = 16; m > 0; m >>= 1) v += __shfl_xor(v, m, 32);
  return v;
}

__device__ __forceinline__ float group8ReduceSum(float v) {
#pragma unroll
  for (int m = 1; m < 8; m <<= 1) v += __shfl_xor(v, m, 32);
  return v;
}

// float atomic max via the classic sign-aware integer punning trick.
__device__ __forceinline__ void atomicMaxF(float* addr, float val) {
  if (val >= 0.0f) atomicMax((int*)addr, __float_as_int(val));
  else             atomicMin((unsigned int*)addr, __float_as_uint(val));
}

// ---------------- small elementwise kernels ----------------

__global__ void k_fill(float* __restrict__ p, float v, int n) {
  int i = blockIdx.x * blockDim.x + threadIdx.x;
  if (i < n) p[i] = v;
}

__global__ void k_cvt_bf16(const float* __restrict__ s, __bf16* __restrict__ d, int n) {
  int i = blockIdx.x * blockDim.x + threadIdx.x;
  if (i < n) d[i] = (__bf16)s[i];
}

// Convert one 128x128 f32 weight (k-major) into bf16 transposed Wt[n][k],
// so WMMA B-fragments become two contiguous per-lane 16B loads.
__global__ void k_wt_bf16(const float* __restrict__ w, __bf16* __restrict__ wt) {
  int idx = blockIdx.x * blockDim.x + threadIdx.x;   // 0..16383
  int k = idx >> 7, n = idx & 127;
  wt[n * DIM + k] = (__bf16)w[k * DIM + n];
}

__global__ void k_add_bias(float* __restrict__ h, const float* __restrict__ b, int n) {
  int i = blockIdx.x * blockDim.x + threadIdx.x;
  if (i < n) h[i] += b[i & (DIM - 1)];
}

// ---------------- LayerNorm + ReLU -> bf16 (GEMM A operand) ----------------

__global__ __launch_bounds__(256)
void k_ln_relu_bf16(const float* __restrict__ h, const float* __restrict__ g,
                    const float* __restrict__ b, __bf16* __restrict__ out) {
  int row  = blockIdx.x * 8 + (threadIdx.x >> 5);
  int lane = threadIdx.x & 31;
  int c = lane * 4;
  float4 v = *(const float4*)(h + (size_t)row * DIM + c);
  float s  = v.x + v.y + v.z + v.w;
  float sq = v.x * v.x + v.y * v.y + v.z * v.z + v.w * v.w;
  s  = waveReduceSum(s);
  sq = waveReduceSum(sq);
  float mu  = s * (1.0f / DIM);
  float var = sq * (1.0f / DIM) - mu * mu;
  float r   = rsqrtf(var + 1e-5f);
  float4 gg = *(const float4*)(g + c);
  float4 bb = *(const float4*)(b + c);
  v4bf o;
  o[0] = (__bf16)fmaxf(0.0f, (v.x - mu) * r * gg.x + bb.x);
  o[1] = (__bf16)fmaxf(0.0f, (v.y - mu) * r * gg.y + bb.y);
  o[2] = (__bf16)fmaxf(0.0f, (v.z - mu) * r * gg.z + bb.z);
  o[3] = (__bf16)fmaxf(0.0f, (v.w - mu) * r * gg.w + bb.w);
  *(v4bf*)(out + (size_t)row * DIM + c) = o;
}

// ---------------- bf16 WMMA GEMM: C[rows x 128] = A[rows x 128] @ W + bias --
// Block = 256 threads = 8 waves; block tile = 16 rows x 128 cols;
// wave w owns cols [16w, 16w+16). K loop: 4 x v_wmma_f32_16x16x32_bf16.

__global__ __launch_bounds__(256)
void k_gemm128(const __bf16* __restrict__ A, const __bf16* __restrict__ Wt,
               const float* __restrict__ bias, float* __restrict__ C) {
  const int wave = threadIdx.x >> 5;
  const int lane = threadIdx.x & 31;
  const int hi   = lane >> 4;          // lane group (K-half select)
  const int m    = lane & 15;          // A row within tile == C/D column-lane
  const int row0 = blockIdx.x * 16;
  const int col  = wave * 16 + m;      // global output column

  const __bf16* arow = A  + (size_t)(row0 + m) * DIM;
  const __bf16* brow = Wt + (size_t)col * DIM;

  v8f acc = {};
#pragma unroll
  for (int kb = 0; kb < DIM; kb += 32) {
    union { v16bf v; v8bf h[2]; } a, b;
    // A 16-bit 16x32 fragment: halfs 0..7 = K kb+hi*8..+7, 8..15 = K kb+16+hi*8..+7
    a.h[0] = *(const v8bf*)(arow + kb + hi * 8);
    a.h[1] = *(const v8bf*)(arow + kb + 16 + hi * 8);
    // B 16-bit 32x16 fragment: halfs 0..7 = K kb+hi*8..+7, 8..15 = K kb+16+hi*8..+7
    b.h[0] = *(const v8bf*)(brow + kb + hi * 8);
    b.h[1] = *(const v8bf*)(brow + kb + 16 + hi * 8);
    acc = __builtin_amdgcn_wmma_f32_16x16x32_bf16(false, a.v, false, b.v,
                                                  (short)0, acc, false, false);
  }
  float bv = bias[col];
#pragma unroll
  for (int r = 0; r < 8; ++r) {                       // VGPR r -> row hi*8 + r
    int row = row0 + hi * 8 + r;
    C[(size_t)row * DIM + col] = acc[r] + bv;
  }
}

// ---------------- edge pass 1: logits + segment max ----------------
// One wave per edge; lane covers 4 channels; head = lane>>3.

__global__ __launch_bounds__(256)
void k_edge_logits(const float* __restrict__ xl, const float* __restrict__ xr,
                   const int* __restrict__ src, const int* __restrict__ dst,
                   const float* __restrict__ ew, const float* __restrict__ We,
                   const float* __restrict__ att, float* __restrict__ logits,
                   float* __restrict__ nodemax) {
  int e = blockIdx.x * 8 + (threadIdx.x >> 5);
  if (e >= NEDGES) return;
  int lane = threadIdx.x & 31;
  int s = src[e], d = dst[e];
  float w = ew[e];
  int c = lane * 4;
  float4 a  = *(const float4*)(xl + (size_t)s * DIM + c);
  float4 b  = *(const float4*)(xr + (size_t)d * DIM + c);
  float4 wv = *(const float4*)(We + c);
  float4 av = *(const float4*)(att + c);   // att is [HEADS][HID] flat = 128
  float t0 = lrelu02(a.x + b.x + w * wv.x);
  float t1 = lrelu02(a.y + b.y + w * wv.y);
  float t2 = lrelu02(a.z + b.z + w * wv.z);
  float t3 = lrelu02(a.w + b.w + w * wv.w);
  float sdot = t0 * av.x + t1 * av.y + t2 * av.z + t3 * av.w;
  sdot = group8ReduceSum(sdot);            // 8 lanes = 32 channels = one head
  if ((lane & 7) == 0) {
    int hh = lane >> 3;
    logits[(size_t)e * HEADS + hh] = sdot;
    atomicMaxF(&nodemax[(size_t)d * HEADS + hh], sdot);
  }
}

// ---------------- edge pass 2: exp + segment sum (in-place ex) -------------

__global__ void k_edge_exp(float* __restrict__ logits, const int* __restrict__ dst,
                           const float* __restrict__ nodemax, float* __restrict__ denom) {
  int idx = blockIdx.x * blockDim.x + threadIdx.x;
  if (idx >= NEDGES * HEADS) return;
  int e = idx >> 2, hh = idx & 3;
  int d = dst[e];
  float v = __expf(logits[idx] - nodemax[(size_t)d * HEADS + hh]);
  logits[idx] = v;                          // reuse buffer as ex
  atomicAdd(&denom[(size_t)d * HEADS + hh], v);
}

// ---------------- edge pass 3: alpha-weighted scatter into h ---------------

__global__ __launch_bounds__(256)
void k_edge_scatter(const float* __restrict__ xl, const int* __restrict__ src,
                    const int* __restrict__ dst, const float* __restrict__ ex,
                    const float* __restrict__ denom, float* __restrict__ h) {
  int e = blockIdx.x * 8 + (threadIdx.x >> 5);
  if (e >= NEDGES) return;
  int lane = threadIdx.x & 31;
  int s = src[e], d = dst[e];
  int c = lane * 4;
  int hh = lane >> 3;
  float alpha = ex[(size_t)e * HEADS + hh] /
                (denom[(size_t)d * HEADS + hh] + 1e-16f);
  float4 a = *(const float4*)(xl + (size_t)s * DIM + c);
  float* hp = h + (size_t)d * DIM + c;
  atomicAdd(hp + 0, a.x * alpha);
  atomicAdd(hp + 1, a.y * alpha);
  atomicAdd(hp + 2, a.z * alpha);
  atomicAdd(hp + 3, a.w * alpha);
}

// ---------------- final: LN + ReLU + 128->1 projection ----------------

__global__ __launch_bounds__(256)
void k_final(const float* __restrict__ h, const float* __restrict__ g,
             const float* __restrict__ b, const float* __restrict__ fcW,
             const float* __restrict__ fcb, float* __restrict__ out) {
  int row  = blockIdx.x * 8 + (threadIdx.x >> 5);
  int lane = threadIdx.x & 31;
  int c = lane * 4;
  float4 v = *(const float4*)(h + (size_t)row * DIM + c);
  float s  = v.x + v.y + v.z + v.w;
  float sq = v.x * v.x + v.y * v.y + v.z * v.z + v.w * v.w;
  s  = waveReduceSum(s);
  sq = waveReduceSum(sq);
  float mu  = s * (1.0f / DIM);
  float var = sq * (1.0f / DIM) - mu * mu;
  float r   = rsqrtf(var + 1e-5f);
  float4 gg = *(const float4*)(g + c);
  float4 bb = *(const float4*)(b + c);
  float4 fw = *(const float4*)(fcW + c);
  float dot = fmaxf(0.0f, (v.x - mu) * r * gg.x + bb.x) * fw.x
            + fmaxf(0.0f, (v.y - mu) * r * gg.y + bb.y) * fw.y
            + fmaxf(0.0f, (v.z - mu) * r * gg.z + bb.z) * fw.z
            + fmaxf(0.0f, (v.w - mu) * r * gg.w + bb.w) * fw.w;
  dot = waveReduceSum(dot);
  if (lane == 0) out[row] = dot + fcb[0];
}

// ---------------- launcher ----------------

extern "C" void kernel_launch(void* const* d_in, const int* in_sizes, int n_in,
                              void* d_out, int out_size, void* d_ws, size_t ws_size,
                              hipStream_t stream) {
  const float* x    = (const float*)d_in[0];
  const int*   ei   = (const int*)d_in[1];
  const float* ewt  = (const float*)d_in[2];
  const float* encW = (const float*)d_in[3];
  const float* encb = (const float*)d_in[4];
  const float* Wl   = (const float*)d_in[5];
  const float* bl   = (const float*)d_in[6];
  const float* Wr   = (const float*)d_in[7];
  const float* br   = (const float*)d_in[8];
  const float* We   = (const float*)d_in[9];
  const float* att  = (const float*)d_in[10];
  const float* convb= (const float*)d_in[11];
  const float* lng  = (const float*)d_in[12];
  const float* lnb  = (const float*)d_in[13];
  const float* lnfg = (const float*)d_in[14];
  const float* lnfb = (const float*)d_in[15];
  const float* fcW  = (const float*)d_in[16];
  const float* fcb  = (const float*)d_in[17];
  float* out = (float*)d_out;

  const int* src = ei;
  const int* dst = ei + NEDGES;

  // Workspace layout (~208 MB).
  char* ws = (char*)d_ws;
  size_t off = 0;
  float* h      = (float*)(ws + off); off += (size_t)NNODES * DIM * 4;
  float* xl     = (float*)(ws + off); off += (size_t)NNODES * DIM * 4;
  float* xr     = (float*)(ws + off); off += (size_t)NNODES * DIM * 4;
  float* logits = (float*)(ws + off); off += (size_t)NEDGES * HEADS * 4;  // reused as ex
  float* nodemax= (float*)(ws + off); off += (size_t)NNODES * HEADS * 4;
  float* denom  = (float*)(ws + off); off += (size_t)NNODES * HEADS * 4;
  __bf16* gbf   = (__bf16*)(ws + off); off += (size_t)NNODES * DIM * 2;   // bf16 A operand
  __bf16* wbf   = (__bf16*)(ws + off); off += (size_t)7 * DIM * DIM * 2;  // transposed bf16 weights

  const int NV = NNODES * DIM;                 // 12.8M node-feature elements
  dim3 blk(256);

  // Weights -> bf16, transposed: [enc, Wl0..2, Wr0..2]
  k_wt_bf16<<<64, blk, 0, stream>>>(encW, wbf);
  for (int l = 0; l < NLAYER; ++l) {
    k_wt_bf16<<<64, blk, 0, stream>>>(Wl + (size_t)l * DIM * DIM, wbf + (size_t)(1 + l) * DIM * DIM);
    k_wt_bf16<<<64, blk, 0, stream>>>(Wr + (size_t)l * DIM * DIM, wbf + (size_t)(4 + l) * DIM * DIM);
  }

  // Encoder: h = x @ enc_W + enc_b  (x -> bf16 first)
  k_cvt_bf16<<<NV / 256, blk, 0, stream>>>(x, gbf, NV);
  k_gemm128<<<NNODES / 16, blk, 0, stream>>>(gbf, wbf, encb, h);

  for (int l = 0; l < NLAYER; ++l) {
    // g = relu(LN(h)) as bf16
    k_ln_relu_bf16<<<NNODES / 8, blk, 0, stream>>>(h, lng + l * DIM, lnb + l * DIM, gbf);
    // xl = g@Wl+bl ; xr = g@Wr+br  (WMMA)
    k_gemm128<<<NNODES / 16, blk, 0, stream>>>(gbf, wbf + (size_t)(1 + l) * DIM * DIM, bl + l * DIM, xl);
    k_gemm128<<<NNODES / 16, blk, 0, stream>>>(gbf, wbf + (size_t)(4 + l) * DIM * DIM, br + l * DIM, xr);
    // segment softmax state
    k_fill<<<(NNODES * HEADS + 255) / 256, blk, 0, stream>>>(nodemax, -INFINITY, NNODES * HEADS);
    k_fill<<<(NNODES * HEADS + 255) / 256, blk, 0, stream>>>(denom, 0.0f, NNODES * HEADS);
    // edge passes
    k_edge_logits<<<NEDGES / 8, blk, 0, stream>>>(xl, xr, src, dst, ewt,
                                                  We + l * DIM, att + l * DIM,
                                                  logits, nodemax);
    k_edge_exp<<<(NEDGES * HEADS) / 256, blk, 0, stream>>>(logits, dst, nodemax, denom);
    // residual: accumulate alpha*xl[src] straight into h
    k_edge_scatter<<<NEDGES / 8, blk, 0, stream>>>(xl, src, dst, logits, denom, h);
    // + conv bias
    k_add_bias<<<NV / 256, blk, 0, stream>>>(h, convb + l * DIM, NV);
  }

  // out = relu(LN(h)) @ fc_W + fc_b
  k_final<<<NNODES / 8, blk, 0, stream>>>(h, lnfg, lnfb, fcW, fcb, out);
}
